// Bilinear_23115513987566
// MI455X (gfx1250) — matrix operand order repaired and checked by reading, not compiled, not run on hardware
//
#include <hip/hip_runtime.h>

// ---------------- CDNA5 vector types ----------------
typedef __attribute__((ext_vector_type(16))) __bf16   v16bf;
typedef __attribute__((ext_vector_type(8)))  float    v8f;
typedef __attribute__((ext_vector_type(4)))  float    v4f;
typedef __attribute__((ext_vector_type(4)))  unsigned v4u;

// ---------------- compile-time PGA(3,0,1) blade algebra ----------------
// Blade order matches reference BLADES list; bit i of mask = basis vector e_i,
// e0 (bit 0) is the degenerate direction (e0^2 = 0).
constexpr int MASK_OF[16] = {0, 1, 2, 4, 8, 3, 5, 9, 6, 10, 12, 7, 11, 13, 14, 15};

constexpr int pc4(int x) { int c = 0; while (x) { c += x & 1; x >>= 1; } return c; }
constexpr int idx_of(int m) { for (int i = 0; i < 16; i++) if (MASK_OF[i] == m) return i; return 0; }
// sign to merge sorted blade a with sorted blade b (count transpositions)
constexpr int reorder(int a, int b) {
  int c = 0; a >>= 1;
  while (a) { c += pc4(a & b); a >>= 1; }
  return (c & 1) ? -1 : 1;
}
constexpr int dual_sign(int m) { return reorder(m, 15 & ~m); } // sign of a ^ comp(a) = e0123

struct Ent { int k, i, j, s; };
struct GPTbl { Ent e[192]; };
constexpr GPTbl make_gp() {
  GPTbl T{}; int n = 0;
  for (int i = 0; i < 16; i++)
    for (int j = 0; j < 16; j++) {
      int mi = MASK_OF[i], mj = MASK_OF[j];
      if (mi & mj & 1) continue;                 // e0*e0 = 0 kills the term
      T.e[n++] = Ent{ idx_of(mi ^ mj), i, j, reorder(mi, mj) };
    }
  return T;
}
struct JNTbl { Ent e[81]; };
constexpr JNTbl make_jn() {
  JNTbl T{}; int n = 0;
  for (int i = 0; i < 16; i++)
    for (int j = 0; j < 16; j++) {
      int mi = MASK_OF[i], mj = MASK_OF[j];
      if ((mi | mj) != 15) continue;             // duals must be disjoint
      int ci = 15 & ~mi, cj = 15 & ~mj;
      int mk = mi & mj;
      int s = dual_sign(mi) * dual_sign(mj) * reorder(ci, cj) * dual_sign(mk);
      T.e[n++] = Ent{ idx_of(mk), i, j, s };
    }
  return T;
}
constexpr GPTbl GPT = make_gp();
constexpr JNTbl JNT = make_jn();

// 24 routed (w-map, input blade b, output blade a) matmul pairs for equi_linear.
// w 0..4: grade projections (a=b); w 5..8: e0 ^ (grade-g, e0-free part), sign +1.
struct PairT { int w[24], b[24], a[24]; };
constexpr PairT make_pairs() {
  PairT P{}; int n = 0;
  for (int w = 0; w < 5; w++)
    for (int b = 0; b < 16; b++)
      if (pc4(MASK_OF[b]) == w) { P.w[n] = w; P.b[n] = b; P.a[n] = b; n++; }
  for (int g = 0; g < 4; g++)
    for (int b = 0; b < 16; b++) {
      int m = MASK_OF[b];
      if ((m & 1) || pc4(m) != g) continue;
      P.w[n] = 5 + g; P.b[n] = b; P.a[n] = idx_of(m | 1); n++;
    }
  return P;
}
constexpr PairT PR = make_pairs();

#define WAVES 8
#define NTHREADS (WAVES * 32)
#define WF_HALFS (54 * 512)           // 36 equi1 + 18 equi2 A-fragment images
#define WAVE_REGION 32768             // 16KB x tile + 16KB y tile (obuf overlays both)

// per-lane sparse bilinear on C/D-layout fragments (blade = fragment idx)
__device__ __forceinline__ void bilinear(int pass, const v8f (&L)[16], const v8f (&R)[16],
                                         v8f (&Y)[16]) {
#pragma unroll
  for (int r = 0; r < 8; r++) {
    float lg[16], rg[16];
#pragma unroll
    for (int i = 0; i < 16; i++) { lg[i] = L[i][r]; rg[i] = R[i][r]; }
    if (pass == 0) {
#pragma unroll
      for (int t = 0; t < 192; t++) {
        const Ent e = GPT.e[t];
        float p = lg[e.i] * rg[e.j];
        Y[e.k][r] = (e.s > 0) ? (Y[e.k][r] + p) : (Y[e.k][r] - p);
      }
    } else {
#pragma unroll
      for (int t = 0; t < 81; t++) {
        const Ent e = JNT.e[t];
        float p = lg[e.i] * rg[e.j];
        Y[e.k][r] = (e.s > 0) ? (Y[e.k][r] + p) : (Y[e.k][r] - p);
      }
    }
  }
}

__global__ __launch_bounds__(NTHREADS)
void pga_fused_kernel(const float* __restrict__ xg, const float* __restrict__ rmg,
                      const float* __restrict__ Wbil, const float* __restrict__ Wout,
                      float* __restrict__ outg, int ntiles) {
  extern __shared__ __align__(32) char smem[];
  __bf16* wf = (__bf16*)smem;
  const int tid = threadIdx.x;
  const int lane = tid & 31;
  const int wave = tid >> 5;
  char* wbase = smem + (size_t)WF_HALFS * 2 + (size_t)wave * WAVE_REGION;
  __bf16* xbuf = (__bf16*)wbase;             // [blade][tok16][ch32] bf16
  __bf16* ybuf = (__bf16*)(wbase + 16384);   // [blade][tok16][ych32] bf16
  float*  obuf = (float*)wbase;              // [tok16][oc32][blade16] f32 (overlay)

  // ---- stage weight A-fragments (pre-swizzled to WMMA A lane layout, bf16) ----
  for (int idx = tid; idx < 54 * 512; idx += NTHREADS) {
    int f = idx >> 9, rem = idx & 511, l = rem >> 4, h = rem & 15;
    int m = l & 15;
    int k = ((l < 16) ? 0 : 16) + h;
    float v;
    if (f < 36) { int w = f >> 2, mc = f & 3; v = Wbil[((mc * 16 + m) * 32 + k) * 9 + w]; }
    else { int f2 = f - 36; int w = f2 >> 1, mc = f2 & 1; v = Wout[((mc * 16 + m) * 32 + k) * 9 + w]; }
    wf[idx] = (__bf16)v;
  }
  __syncthreads();

  const int n = lane & 15, hi = lane >> 4;
  const int nwaves = gridDim.x * WAVES;

  for (int tile = blockIdx.x * WAVES + wave; tile < ntiles; tile += nwaves) {
    const int tb = tile * 16;                       // token base of this tile

    // ---- prefetch next tile's x region into cache (global_prefetch_b8) ----
    if (tile + nwaves < ntiles) {
      const char* pb = (const char*)(xg + (size_t)(tile + nwaves) * 16 * 512);
#pragma unroll
      for (int i = 0; i < 8; i++)
        __builtin_prefetch(pb + (size_t)(i * 32 + lane) * 128, 0, 0);
    }

    // ---- coalesced x tile load -> bf16 LDS in B-fragment-friendly layout ----
    const float* xt = xg + (size_t)tb * 512;
#pragma unroll 4
    for (int i = 0; i < 64; i++) {
      int e0 = (i * 32 + lane) * 4;
      v4f v = *(const v4f*)(xt + e0);
      int tok = e0 >> 9, rem = e0 & 511, ch = rem >> 4, a0 = rem & 15;
#pragma unroll
      for (int q = 0; q < 4; q++)
        xbuf[((a0 + q) * 16 + tok) * 32 + ch] = (__bf16)v[q];
    }
    __builtin_amdgcn_wave_barrier();

    const float rmv = rmg[(size_t)(tb + n) * 16 + 14];

    // ---- equi_linear1 (WMMA) + GP/JOIN bilinear (VALU), two hidden-chunk pairs ----
#pragma unroll
    for (int pass = 0; pass < 2; pass++) {
      v8f accL[16] = {}, accR[16] = {};
#pragma unroll
      for (int p = 0; p < 24; p++) {
        const int w = PR.w[p], b = PR.b[p], a = PR.a[p];
        v16bf Bf = *(const v16bf*)(xbuf + (b * 16 + n) * 32 + hi * 16);
        v16bf AL = *(const v16bf*)(wf + (w * 4 + pass * 2 + 0) * 512 + lane * 16);
        v16bf AR = *(const v16bf*)(wf + (w * 4 + pass * 2 + 1) * 512 + lane * 16);
        accL[a] = __builtin_amdgcn_wmma_f32_16x16x32_bf16(false, AL, false, Bf, (short)0,
                                                          accL[a], false, false);
        accR[a] = __builtin_amdgcn_wmma_f32_16x16x32_bf16(false, AR, false, Bf, (short)0,
                                                          accR[a], false, false);
      }
      v8f yk[16] = {};
      bilinear(pass, accL, accR, yk);
      // packed bf16 store: 8 consecutive y-channels -> one ds_store_b128 per blade
#pragma unroll
      for (int k = 0; k < 16; k++) {
        v4u pk;
#pragma unroll
        for (int rr = 0; rr < 4; rr++) {
          float v0 = yk[k][2 * rr], v1 = yk[k][2 * rr + 1];
          if (pass == 1) { v0 *= rmv; v1 *= rmv; }   // join scaled by ref_mv e123
          unsigned short b0 = __builtin_bit_cast(unsigned short, (__bf16)v0);
          unsigned short b1 = __builtin_bit_cast(unsigned short, (__bf16)v1);
          pk[rr] = (unsigned)b0 | ((unsigned)b1 << 16);
        }
        *(v4u*)((char*)ybuf + ((k * 16 + n) * 64 + pass * 32 + 16 * hi)) = pk;
      }
      __builtin_amdgcn_wave_barrier();
    }

    // ---- equi_linear2 (WMMA), both output chunks ----
    v8f oacc0[16] = {}, oacc1[16] = {};
#pragma unroll
    for (int p = 0; p < 24; p++) {
      const int w = PR.w[p], b = PR.b[p], a = PR.a[p];
      v16bf Yf = *(const v16bf*)(ybuf + (b * 16 + n) * 32 + hi * 16);
      v16bf A0 = *(const v16bf*)(wf + (36 + w * 2 + 0) * 512 + lane * 16);
      v16bf A1 = *(const v16bf*)(wf + (36 + w * 2 + 1) * 512 + lane * 16);
      oacc0[a] = __builtin_amdgcn_wmma_f32_16x16x32_bf16(false, A0, false, Yf, (short)0,
                                                         oacc0[a], false, false);
      oacc1[a] = __builtin_amdgcn_wmma_f32_16x16x32_bf16(false, A1, false, Yf, (short)0,
                                                         oacc1[a], false, false);
    }
    __builtin_amdgcn_wave_barrier();

    // ---- packed f32 scatter to obuf: 4 blades -> one ds_store_b128 ----
#pragma unroll
    for (int r = 0; r < 8; r++) {
      const int oc = r + 8 * hi;
      float* row0 = obuf + (n * 32 + oc) * 16;
      float* row1 = obuf + (n * 32 + 16 + oc) * 16;
#pragma unroll
      for (int a4 = 0; a4 < 4; a4++) {
        v4f v0 = { oacc0[a4 * 4 + 0][r], oacc0[a4 * 4 + 1][r],
                   oacc0[a4 * 4 + 2][r], oacc0[a4 * 4 + 3][r] };
        v4f v1 = { oacc1[a4 * 4 + 0][r], oacc1[a4 * 4 + 1][r],
                   oacc1[a4 * 4 + 2][r], oacc1[a4 * 4 + 3][r] };
        *(v4f*)(row0 + a4 * 4) = v0;
        *(v4f*)(row1 + a4 * 4) = v1;
      }
    }
    __builtin_amdgcn_wave_barrier();

    // ---- fully coalesced global writeback from obuf ----
    float* og = outg + (size_t)tb * 512;
#pragma unroll 4
    for (int i = 0; i < 64; i++) {
      int e0 = (i * 32 + lane) * 4;
      *(v4f*)(og + e0) = *(const v4f*)(obuf + e0);
    }
    __builtin_amdgcn_wave_barrier();
  }
}

extern "C" void kernel_launch(void* const* d_in, const int* in_sizes, int n_in,
                              void* d_out, int out_size, void* d_ws, size_t ws_size,
                              hipStream_t stream) {
  const float* x  = (const float*)d_in[0];
  const float* rm = (const float*)d_in[1];
  const float* wb = (const float*)d_in[2];
  const float* wo = (const float*)d_in[3];
  float* out = (float*)d_out;
  (void)d_ws; (void)ws_size; (void)n_in; (void)out_size;

  int tokens = in_sizes[0] / 512;           // B*S (x is [tokens, 32, 16] f32)
  int ntiles = tokens / 16;                 // 16 tokens per wave-tile
  int blocks = ntiles / (WAVES * 2);        // 2 tiles per wave (grid-stride + prefetch)
  if (blocks < 1) blocks = 1;
  size_t shmem = (size_t)WF_HALFS * 2 + (size_t)WAVES * WAVE_REGION;  // 317440 B

  hipFuncSetAttribute((const void*)pga_fused_kernel,
                      hipFuncAttributeMaxDynamicSharedMemorySize, (int)shmem);
  pga_fused_kernel<<<blocks, NTHREADS, shmem, stream>>>(x, rm, wb, wo, out, ntiles);
}
